// DepthwiseXCorr_2843268350207
// MI455X (gfx1250) — matrix operand order repaired and checked by reading, not compiled, not run on hardware
//
#include <hip/hip_runtime.h>

typedef __attribute__((ext_vector_type(16))) __bf16 v16bf;
typedef __attribute__((ext_vector_type(8)))  float  v8f;
typedef __attribute__((ext_vector_type(8)))  unsigned short u16x8;

union FragBF {
    v16bf v;
    u16x8 u[2];
};

__device__ __forceinline__ unsigned short f2bf(float f) {
    unsigned int u = __float_as_uint(f);
    u += 0x7FFFu + ((u >> 16) & 1u);   // round-to-nearest-even
    return (unsigned short)(u >> 16);
}

// ---------------------------------------------------------------------------
// Prep: fold BN scale into weights, convert to bf16 [N][K] layout, compute betas
// ---------------------------------------------------------------------------
__global__ __launch_bounds__(256)
void prep_kernel(const float* __restrict__ wk,
                 const float* __restrict__ ksc, const float* __restrict__ kbi,
                 const float* __restrict__ kme, const float* __restrict__ kva,
                 const float* __restrict__ wsw,
                 const float* __restrict__ ssc, const float* __restrict__ sbi,
                 const float* __restrict__ sme, const float* __restrict__ sva,
                 const float* __restrict__ w1,
                 const float* __restrict__ hsc, const float* __restrict__ hbi,
                 const float* __restrict__ hme, const float* __restrict__ hva,
                 const float* __restrict__ w2,
                 unsigned short* __restrict__ wkB, unsigned short* __restrict__ wsB,
                 unsigned short* __restrict__ w1B, unsigned short* __restrict__ w2B,
                 float* __restrict__ betak, float* __restrict__ betas,
                 float* __restrict__ betah)
{
    int i = blockIdx.x * blockDim.x + threadIdx.x;
    if (i < 2304 * 256) {                 // wk / ws : [O=256][K=2304] flat
        int o = i / 2304;
        float invk = ksc[o] * rsqrtf(kva[o] + 1e-5f);
        wkB[i] = f2bf(wk[i] * invk);
        float invs = ssc[o] * rsqrtf(sva[o] + 1e-5f);
        wsB[i] = f2bf(wsw[i] * invs);
    }
    if (i < 256 * 256) {                  // w1 : [256][256]
        int o = i / 256;
        float invh = hsc[o] * rsqrtf(hva[o] + 1e-5f);
        w1B[i] = f2bf(w1[i] * invh);
    }
    if (i < 32 * 256) {                   // w2 : [20][256] padded to 32 rows
        int o = i / 256;
        w2B[i] = (o < 20) ? f2bf(w2[i]) : (unsigned short)0;
    }
    if (i < 256) {
        float invk = ksc[i] * rsqrtf(kva[i] + 1e-5f);
        betak[i] = kbi[i] - kme[i] * invk;
        float invs = ssc[i] * rsqrtf(sva[i] + 1e-5f);
        betas[i] = sbi[i] - sme[i] * invs;
        float invh = hsc[i] * rsqrtf(hva[i] + 1e-5f);
        betah[i] = hbi[i] - hme[i] * invh;
    }
}

// ---------------------------------------------------------------------------
// 3x3 conv (VALID) as implicit GEMM with v_wmma_f32_16x16x32_bf16.
// Block tile: M=32 output pixels x N=256 out channels, K = 2304 (=256*9).
// 256 threads = 8 waves; each wave owns n-tiles {wave, wave+8} and both
// 16-row M-subtiles -> 4 accumulators, 4 WMMAs per K-chunk from
// 2 A-fragments (LDS) + 2 B-fragments (global [N][K] bf16).
// BN folded into weights; epilogue adds beta + ReLU; output NCHW f32.
// ---------------------------------------------------------------------------
#define CONV_KTOT 2304
#define CONV_KSUP 256
#define CONV_BM   32

__global__ __launch_bounds__(256)
void conv3x3_wmma(const float* __restrict__ x, int Hin, int Win, int Wout, int Mtot,
                  const unsigned short* __restrict__ wB,
                  const float* __restrict__ beta,
                  float* __restrict__ y /* [B][256][Mtot] */)
{
    __shared__ unsigned short Alds[CONV_BM][CONV_KSUP + 8];

    const int b      = blockIdx.y;
    const int m_base = blockIdx.x * CONV_BM;
    const int tid    = threadIdx.x;
    const int lane   = tid & 31;
    const int wave   = tid >> 5;       // 0..7
    const int mrow   = lane & 15;
    const int hi     = lane >> 4;      // half-wave select
    const int nloc   = lane & 15;

    const float* xb = x + (size_t)b * 256 * Hin * Win;

    v8f acc00 = {};   // (m-sub 0, n-tile 0)
    v8f acc01 = {};   // (m-sub 0, n-tile 1)
    v8f acc10 = {};   // (m-sub 1, n-tile 0)
    v8f acc11 = {};   // (m-sub 1, n-tile 1)

    for (int ks = 0; ks < CONV_KTOT; ks += CONV_KSUP) {
        __syncthreads();
        // --- stage A (im2col patch) 32 x 256 into LDS as bf16 ---
        for (int e = tid; e < CONV_BM * CONV_KSUP; e += 256) {
            int m  = e >> 8;          // e / 256
            int kk = e & 255;
            int k  = ks + kk;
            int p  = m_base + m;
            unsigned short val = 0;
            if (p < Mtot) {
                int oy = p / Wout, ox = p % Wout;
                int ci = k / 9, r = k % 9;
                int ky = r / 3, kx = r % 3;
                val = f2bf(xb[((size_t)ci * Hin + (oy + ky)) * Win + (ox + kx)]);
            }
            Alds[m][kk] = val;
        }
        __syncthreads();

        // --- 8 WMMA K-chunks over this super-chunk ---
        for (int kc = 0; kc < CONV_KSUP; kc += 32) {
            FragBF a0, a1;
            a0.u[0] = *(const u16x8*)&Alds[mrow][kc + hi * 16];
            a0.u[1] = *(const u16x8*)&Alds[mrow][kc + hi * 16 + 8];
            a1.u[0] = *(const u16x8*)&Alds[16 + mrow][kc + hi * 16];
            a1.u[1] = *(const u16x8*)&Alds[16 + mrow][kc + hi * 16 + 8];

            const int kg = ks + kc + hi * 16;
            const unsigned short* bp0 =
                wB + (size_t)(wave * 16 + nloc) * CONV_KTOT + kg;
            const unsigned short* bp1 =
                wB + (size_t)((wave + 8) * 16 + nloc) * CONV_KTOT + kg;
            FragBF b0, b1;
            b0.u[0] = *(const u16x8*)(bp0);
            b0.u[1] = *(const u16x8*)(bp0 + 8);
            b1.u[0] = *(const u16x8*)(bp1);
            b1.u[1] = *(const u16x8*)(bp1 + 8);

            acc00 = __builtin_amdgcn_wmma_f32_16x16x32_bf16(
                false, a0.v, false, b0.v, (short)0, acc00, false, false);
            acc10 = __builtin_amdgcn_wmma_f32_16x16x32_bf16(
                false, a1.v, false, b0.v, (short)0, acc10, false, false);
            acc01 = __builtin_amdgcn_wmma_f32_16x16x32_bf16(
                false, a0.v, false, b1.v, (short)0, acc01, false, false);
            acc11 = __builtin_amdgcn_wmma_f32_16x16x32_bf16(
                false, a1.v, false, b1.v, (short)0, acc11, false, false);
        }
    }

    // --- epilogue: + beta, ReLU, store NCHW f32 ---
    const int n0 = wave * 16 + nloc;
    const int n1 = (wave + 8) * 16 + nloc;
    const float bn0 = beta[n0];
    const float bn1 = beta[n1];
    float* y0 = y + ((size_t)b * 256 + n0) * Mtot;
    float* y1 = y + ((size_t)b * 256 + n1) * Mtot;

    #pragma unroll
    for (int r = 0; r < 8; ++r) {
        int p0 = m_base + r + hi * 8;        // m-sub 0
        int p1 = p0 + 16;                     // m-sub 1
        if (p0 < Mtot) {
            float v = acc00[r] + bn0;
            y0[p0] = v > 0.f ? v : 0.f;
            v = acc01[r] + bn1;
            y1[p0] = v > 0.f ? v : 0.f;
        }
        if (p1 < Mtot) {
            float v = acc10[r] + bn0;
            y0[p1] = v > 0.f ? v : 0.f;
            v = acc11[r] + bn1;
            y1[p1] = v > 0.f ? v : 0.f;
        }
    }
}

// ---------------------------------------------------------------------------
// Per-sample depthwise cross-correlation (tiny, 0.5 GFLOP) — plain f32 VALU.
// Emits feature map as bf16 in [b][pixel][channel] layout for the head GEMM.
// ---------------------------------------------------------------------------
__global__ __launch_bounds__(256)
void xcorr_kernel(const float* __restrict__ s,   // [64][256][29*29]
                  const float* __restrict__ kf,  // [64][256][5*5]
                  unsigned short* __restrict__ featB) // [64][625][256] bf16
{
    int idx = blockIdx.x * blockDim.x + threadIdx.x;
    if (idx >= 64 * 256 * 625) return;
    int p = idx % 625;
    int c = (idx / 625) & 255;
    int b = idx / (625 * 256);
    int oy = p / 25, ox = p % 25;
    const float* sp = s  + ((size_t)(b * 256 + c) * 29 + oy) * 29 + ox;
    const float* kp = kf + (size_t)(b * 256 + c) * 25;
    float sum = 0.f;
    #pragma unroll
    for (int ky = 0; ky < 5; ++ky)
        #pragma unroll
        for (int kx = 0; kx < 5; ++kx)
            sum += sp[ky * 29 + kx] * kp[ky * 5 + kx];
    featB[((size_t)b * 625 + p) * 256 + c] = f2bf(sum);
}

// ---------------------------------------------------------------------------
// Head: (1x1 conv 256->256, BN, ReLU) then (1x1 conv 256->20 + bias).
// One wave per 16-pixel tile. GEMM1 result held as bf16 tile in LDS, feeds GEMM2.
// ---------------------------------------------------------------------------
__global__ __launch_bounds__(32)
void head_wmma(const unsigned short* __restrict__ featB, // [64][625][256] bf16
               const unsigned short* __restrict__ w1B,   // [256][256] bf16 (BN folded)
               const float* __restrict__ betah,
               const unsigned short* __restrict__ w2B,   // [32][256] bf16 (rows>=20 zero)
               const float* __restrict__ b2,
               float* __restrict__ out)                  // [64][20][625] f32
{
    __shared__ unsigned short Hlds[16][256 + 8];

    const int b      = blockIdx.y;
    const int m_base = blockIdx.x * 16;    // 40 tiles over 625 pixels
    const int lane   = threadIdx.x & 31;
    const int mrow   = lane & 15;
    const int hi     = lane >> 4;
    const int nloc   = lane & 15;

    int prow = m_base + mrow;
    if (prow > 624) prow = 624;            // clamp A reads; extra rows unused
    const unsigned short* arow = featB + ((size_t)b * 625 + prow) * 256;

    // ---- GEMM1: h = relu(w1'·feat + betah), bf16 tile into LDS ----
    for (int nt = 0; nt < 16; ++nt) {
        v8f acc = {};
        for (int kc = 0; kc < 256; kc += 32) {
            FragBF a;
            a.u[0] = *(const u16x8*)(arow + kc + hi * 16);
            a.u[1] = *(const u16x8*)(arow + kc + hi * 16 + 8);
            const unsigned short* bp = w1B + (size_t)(nt * 16 + nloc) * 256 + kc + hi * 16;
            FragBF bb;
            bb.u[0] = *(const u16x8*)(bp);
            bb.u[1] = *(const u16x8*)(bp + 8);
            acc = __builtin_amdgcn_wmma_f32_16x16x32_bf16(
                false, a.v, false, bb.v, (short)0, acc, false, false);
        }
        float bn = betah[nt * 16 + nloc];
        #pragma unroll
        for (int r = 0; r < 8; ++r) {
            float v = acc[r] + bn;
            Hlds[r + hi * 8][nt * 16 + nloc] = f2bf(v > 0.f ? v : 0.f);
        }
    }
    __syncthreads();

    // ---- GEMM2: out = w2·h + b2 (N padded to 32) ----
    for (int nt = 0; nt < 2; ++nt) {
        v8f acc = {};
        for (int kc = 0; kc < 256; kc += 32) {
            FragBF a;
            a.u[0] = *(const u16x8*)&Hlds[mrow][kc + hi * 16];
            a.u[1] = *(const u16x8*)&Hlds[mrow][kc + hi * 16 + 8];
            const unsigned short* bp = w2B + (size_t)(nt * 16 + nloc) * 256 + kc + hi * 16;
            FragBF bb;
            bb.u[0] = *(const u16x8*)(bp);
            bb.u[1] = *(const u16x8*)(bp + 8);
            acc = __builtin_amdgcn_wmma_f32_16x16x32_bf16(
                false, a.v, false, bb.v, (short)0, acc, false, false);
        }
        int n = nt * 16 + nloc;
        if (n < 20) {
            float bias = b2[n];
            #pragma unroll
            for (int r = 0; r < 8; ++r) {
                int p = m_base + r + hi * 8;
                if (p < 625)
                    out[((size_t)b * 20 + n) * 625 + p] = acc[r] + bias;
            }
        }
    }
}

// ---------------------------------------------------------------------------
extern "C" void kernel_launch(void* const* d_in, const int* in_sizes, int n_in,
                              void* d_out, int out_size, void* d_ws, size_t ws_size,
                              hipStream_t stream)
{
    const float* kernel_in = (const float*)d_in[0];   // [64,256,7,7]
    const float* search_in = (const float*)d_in[1];   // [64,256,31,31]
    const float* wk        = (const float*)d_in[2];
    const float* bnk_sc    = (const float*)d_in[3];
    const float* bnk_bi    = (const float*)d_in[4];
    const float* bnk_me    = (const float*)d_in[5];
    const float* bnk_va    = (const float*)d_in[6];
    const float* wsw       = (const float*)d_in[7];
    const float* bns_sc    = (const float*)d_in[8];
    const float* bns_bi    = (const float*)d_in[9];
    const float* bns_me    = (const float*)d_in[10];
    const float* bns_va    = (const float*)d_in[11];
    const float* w1        = (const float*)d_in[12];
    const float* bnh_sc    = (const float*)d_in[13];
    const float* bnh_bi    = (const float*)d_in[14];
    const float* bnh_me    = (const float*)d_in[15];
    const float* bnh_va    = (const float*)d_in[16];
    const float* w2        = (const float*)d_in[17];
    const float* b2        = (const float*)d_in[18];
    float* out             = (float*)d_out;           // [64,20,25,25]

    char* ws = (char*)d_ws;
    unsigned short* wkB   = (unsigned short*)(ws + 0);         // 1,179,648 B
    unsigned short* wsB   = (unsigned short*)(ws + 1179648);   // 1,179,648 B
    unsigned short* w1B   = (unsigned short*)(ws + 2359296);   //   131,072 B
    unsigned short* w2B   = (unsigned short*)(ws + 2490368);   //    16,384 B
    float*          betak = (float*)(ws + 2506752);            //     1,024 B
    float*          betas = (float*)(ws + 2507776);
    float*          betah = (float*)(ws + 2508800);
    float*          kfeat = (float*)(ws + 2509824);            // 1,638,400 B  [64][256][25]
    float*          sfeat = (float*)(ws + 4148224);            // 55,115,776 B [64][256][841]
    unsigned short* featB = (unsigned short*)(ws + 59264000);  // 20,480,000 B [64][625][256]

    // 1) fold BN into bf16 weights + betas
    prep_kernel<<<dim3((2304 * 256 + 255) / 256), 256, 0, stream>>>(
        wk, bnk_sc, bnk_bi, bnk_me, bnk_va,
        wsw, bns_sc, bns_bi, bns_me, bns_va,
        w1, bnh_sc, bnh_bi, bnh_me, bnh_va, w2,
        wkB, wsB, w1B, w2B, betak, betas, betah);

    // 2) template branch: 7x7 -> 5x5 (Mtot=25, 1 m-tile of 32)
    conv3x3_wmma<<<dim3(1, 64), 256, 0, stream>>>(
        kernel_in, 7, 7, 5, 25, wkB, betak, kfeat);

    // 3) search branch: 31x31 -> 29x29 (Mtot=841, 27 m-tiles of 32)
    conv3x3_wmma<<<dim3(27, 64), 256, 0, stream>>>(
        search_in, 31, 31, 29, 841, wsB, betas, sfeat);

    // 4) depthwise xcorr -> bf16 [b][pixel][channel]
    xcorr_kernel<<<dim3((64 * 256 * 625 + 255) / 256), 256, 0, stream>>>(
        sfeat, kfeat, featB);

    // 5) head GEMMs -> output
    head_wmma<<<dim3(40, 64), 32, 0, stream>>>(
        featB, w1B, betah, w2B, b2, out);
}
